// Dual_net_47227460387442
// MI455X (gfx1250) — matrix operand order repaired and checked by reading, not compiled, not run on hardware
//
#include <hip/hip_runtime.h>

// CDNA5 / gfx1250, wave32. bf16 WMMA 16x16x32 with f32 accumulation.

typedef __attribute__((ext_vector_type(16))) __bf16 v16bf;
typedef __attribute__((ext_vector_type(8)))  __bf16 v8bf;
typedef __attribute__((ext_vector_type(8)))  float  v8f;

// -------- bf16 workspace layout (element offsets), weights pre-swizzled to
// WMMA B-operand lane order: block(kc,nt) -> 32 lanes x 16 contiguous bf16.
#define OFF_CW1 0        // 32x128   (K=4   zero-padded)
#define OFF_CW2 4096     // 128x16   (N=3   zero-padded)
#define OFF_NW1 6144     // 64x128   (K=54  zero-padded)
#define OFF_NW2 14336    // 128x64   (N=54  zero-padded)
#define OFF_FW1 22528    // 64x128   (K=57  zero-padded)
#define OFF_FW2 30720    // 128x64   (N=57  zero-padded)
#define OFF_OW1 38912    // 54 x (64x128)
#define OFF_OW2 481280   // 54 x (128x16)
#define WS_ELEMS 591872  // ~1.13 MB bf16

// Convert f32 row-major [K][N] (batched) weights into zero-padded bf16 blocks
// in B-operand lane layout:
//   dst[((b*KC*NT + kc*NT + nt)*32 + lane)*16 + e]
//   lane<16 : e0..7 -> K = kc*32 + e,     e8..15 -> K = kc*32 + 8 + e,  N = nt*16 + lane
//   lane>=16: e0..7 -> K = kc*32 + 8 + e, e8..15 -> K = kc*32 + 16 + e, N = nt*16 + (lane-16)
__global__ void prep_weights(const float* __restrict__ src, __bf16* __restrict__ dst,
                             int Ksrc, int Nsrc, int KC, int NT, int nBatch)
{
    int perB  = KC * NT * 512;
    int total = perB * nBatch;
    int idx   = blockIdx.x * 256 + threadIdx.x;
    if (idx >= total) return;
    int b    = idx / perB;
    int r    = idx - b * perB;
    int blk  = r >> 9;
    int lane = (r >> 4) & 31;
    int e    = r & 15;
    int kc   = blk / NT;
    int nt   = blk - kc * NT;
    int k    = kc * 32 + ((e < 8) ? (((lane >> 4) << 3) + e) : (8 + ((lane >> 4) << 3) + e));
    int n    = nt * 16 + (lane & 15);
    float v  = 0.0f;
    if (k < Ksrc && n < Nsrc) v = src[(b * Ksrc + k) * Nsrc + n];
    dst[idx] = (__bf16)v;
}

__device__ __forceinline__ v8f wmma_bf16(v16bf a, v16bf b, v8f c) {
    return __builtin_amdgcn_wmma_f32_16x16x32_bf16(false, a, false, b, (short)0, c,
                                                   false, false);
}

// wave-internal LDS write->cross-lane-read ordering (DS ops are in-order per
// wave; only completion visibility needs the wait).
__device__ __forceinline__ void lds_fence() {
    asm volatile("s_wait_dscnt 0" ::: "memory");
}

// A operand (16xK row-major bf16 in LDS) -> 16-bit A VGPR layout:
//   lane<16 : row=lane,    e0..7 -> K=kc*32+0..7,  e8..15 -> K=kc*32+16..23
//   lane>=16: row=lane-16, e0..7 -> K=kc*32+8..15, e8..15 -> K=kc*32+24..31
__device__ __forceinline__ v16bf load_a(const __bf16* s, int lane, int kc, int kpad) {
    int row  = lane & 15;
    int hsel = (lane < 16) ? 0 : 8;
    int off  = row * kpad + kc * 32 + hsel;
    v8bf lo = *(const v8bf*)(s + off);
    v8bf hi = *(const v8bf*)(s + off + 16);
    return __builtin_shufflevector(lo, hi, 0, 1, 2, 3, 4, 5, 6, 7,
                                           8, 9, 10, 11, 12, 13, 14, 15);
}

// B operand: pre-swizzled workspace block, 32 contiguous bytes per lane.
__device__ __forceinline__ v16bf load_b(const __bf16* w, int lane) {
    return *(const v16bf*)(w + lane * 16);
}

// D layout (lane = column within 16-tile, VGPR r = row r + (lane<16?0:8)) ->
// bf16 row-major LDS store with bias (+ optional relu).
__device__ __forceinline__ void store_acc(__bf16* dst, int stride, int lane, int col,
                                          v8f acc, float bias, bool relu, bool valid) {
    if (valid) {
        int rb = (lane < 16) ? 0 : 8;
        #pragma unroll
        for (int r = 0; r < 8; ++r) {
            float v = acc[r] + bias;
            if (relu) v = fmaxf(v, 0.0f);
            dst[(rb + r) * stride + col] = (__bf16)v;
        }
    }
}

__global__ __launch_bounds__(256) void dualnet_wmma(
    const float* __restrict__ x, const __bf16* __restrict__ ws,
    const float* __restrict__ Cb1, const float* __restrict__ Cb2,
    const float* __restrict__ Nb1, const float* __restrict__ Nb2,
    const float* __restrict__ Fb1, const float* __restrict__ Fb2,
    const float* __restrict__ Ob1, const float* __restrict__ Ob2,
    float* __restrict__ out, int rows)
{
    __shared__ __bf16 stg_all[8][16 * 128];  // per-wave activation staging (A source)
    __shared__ __bf16 fb_all[8][16 * 64];    // per-wave fused-feature buffer
    __shared__ float  sm_all[8][16 * 4];     // per-wave logits for softmax

    const int wave = threadIdx.x >> 5;
    const int lane = threadIdx.x & 31;
    const int colq = lane & 15;              // column within a 16-wide D tile
    const int row0 = (blockIdx.x * 8 + wave) * 16;

    __bf16* stg = stg_all[wave];
    __bf16* fb  = fb_all[wave];
    float*  sm  = sm_all[wave];

    const __bf16* wC1 = ws + OFF_CW1;
    const __bf16* wC2 = ws + OFF_CW2;
    const __bf16* wN1 = ws + OFF_NW1;
    const __bf16* wN2 = ws + OFF_NW2;
    const __bf16* wF1 = ws + OFF_FW1;
    const __bf16* wF2 = ws + OFF_FW2;
    const __bf16* wO1 = ws + OFF_OW1;
    const __bf16* wO2 = ws + OFF_OW2;

    // ---- stage xc (16x32): cols 0..3 = x[:, 0:4], rest zero -----------------
    #pragma unroll
    for (int j = 0; j < 16; ++j) {
        int t = lane * 16 + j;
        int r = t >> 5, c = t & 31;
        int rr = row0 + r; if (rr >= rows) rr = rows - 1;
        float v = (c < 4) ? x[rr * 58 + c] : 0.0f;
        stg[r * 32 + c] = (__bf16)v;
    }
    // zero the f buffer (cols 57..63 must stay zero as K padding)
    #pragma unroll
    for (int j = 0; j < 32; ++j)
        fb[lane * 32 + j] = (__bf16)0.0f;
    lds_fence();

    // ---- C layer 1: [16x32] x [32x128] -> relu -> stg (16x128) --------------
    {
        v16bf a = load_a(stg, lane, 0, 32);
        #pragma unroll
        for (int nt = 0; nt < 8; ++nt) {
            v8f acc = {0, 0, 0, 0, 0, 0, 0, 0};
            acc = wmma_bf16(a, load_b(wC1 + nt * 512, lane), acc);
            int col = nt * 16 + colq;
            store_acc(stg, 128, lane, col, acc, Cb1[col], true, true);
        }
    }
    lds_fence();
    // ---- C layer 2: [16x128] x [128x16] -> fb cols 0..2 ---------------------
    {
        v8f acc = {0, 0, 0, 0, 0, 0, 0, 0};
        #pragma unroll
        for (int kc = 0; kc < 4; ++kc)
            acc = wmma_bf16(load_a(stg, lane, kc, 128), load_b(wC2 + kc * 512, lane), acc);
        bool v = colq < 3;
        store_acc(fb, 64, lane, colq, acc, v ? Cb2[colq] : 0.0f, false, v);
    }
    // ---- stage xn (16x64): cols 0..53 = x[:, 4:58], rest zero ---------------
    #pragma unroll
    for (int j = 0; j < 32; ++j) {
        int t = lane * 32 + j;
        int r = t >> 6, c = t & 63;
        int rr = row0 + r; if (rr >= rows) rr = rows - 1;
        float v = (c < 54) ? x[rr * 58 + 4 + c] : 0.0f;
        stg[r * 64 + c] = (__bf16)v;
    }
    lds_fence();
    // ---- N layer 1: [16x64] x [64x128] -> relu -> stg -----------------------
    {
        v16bf a0 = load_a(stg, lane, 0, 64);
        v16bf a1 = load_a(stg, lane, 1, 64);
        #pragma unroll
        for (int nt = 0; nt < 8; ++nt) {
            v8f acc = {0, 0, 0, 0, 0, 0, 0, 0};
            acc = wmma_bf16(a0, load_b(wN1 + nt * 512, lane), acc);
            acc = wmma_bf16(a1, load_b(wN1 + (8 + nt) * 512, lane), acc);
            int col = nt * 16 + colq;
            store_acc(stg, 128, lane, col, acc, Nb1[col], true, true);
        }
    }
    lds_fence();
    // ---- N layer 2: [16x128] x [128x64] -> fb cols 3..56 --------------------
    #pragma unroll
    for (int nt = 0; nt < 4; ++nt) {
        v8f acc = {0, 0, 0, 0, 0, 0, 0, 0};
        #pragma unroll
        for (int kc = 0; kc < 4; ++kc)
            acc = wmma_bf16(load_a(stg, lane, kc, 128), load_b(wN2 + (kc * 4 + nt) * 512, lane), acc);
        int col = nt * 16 + colq;
        bool v = col < 54;
        store_acc(fb, 64, lane, 3 + col, acc, v ? Nb2[col] : 0.0f, false, v);
    }
    lds_fence();
    // ---- F layer 1: [16x64](fb) x [64x128] -> relu -> stg -------------------
    {
        v16bf a0 = load_a(fb, lane, 0, 64);
        v16bf a1 = load_a(fb, lane, 1, 64);
        #pragma unroll
        for (int nt = 0; nt < 8; ++nt) {
            v8f acc = {0, 0, 0, 0, 0, 0, 0, 0};
            acc = wmma_bf16(a0, load_b(wF1 + nt * 512, lane), acc);
            acc = wmma_bf16(a1, load_b(wF1 + (8 + nt) * 512, lane), acc);
            int col = nt * 16 + colq;
            store_acc(stg, 128, lane, col, acc, Fb1[col], true, true);
        }
    }
    lds_fence();
    // ---- F layer 2: [16x128] x [128x64] -> fb cols 0..56 --------------------
    #pragma unroll
    for (int nt = 0; nt < 4; ++nt) {
        v8f acc = {0, 0, 0, 0, 0, 0, 0, 0};
        #pragma unroll
        for (int kc = 0; kc < 4; ++kc)
            acc = wmma_bf16(load_a(stg, lane, kc, 128), load_b(wF2 + (kc * 4 + nt) * 512, lane), acc);
        int col = nt * 16 + colq;
        bool v = col < 57;
        store_acc(fb, 64, lane, col, acc, v ? Fb2[col] : 0.0f, false, v);
    }
    lds_fence();

    // ---- 54 output branches: h = relu(f @ Ow1[n] + b1); logits = h @ Ow2[n] -
    v16bf g0 = load_a(fb, lane, 0, 64);   // f is invariant across branches
    v16bf g1 = load_a(fb, lane, 1, 64);
    for (int n = 0; n < 54; ++n) {
        const __bf16* w1 = wO1 + n * 8192;
        const __bf16* w2 = wO2 + n * 2048;
        if (n + 1 < 54) {  // warm next branch's weights into WGP$/L2
            __builtin_prefetch(wO1 + (n + 1) * 8192 + lane * 256, 0, 1);
            __builtin_prefetch(wO2 + (n + 1) * 2048 + lane * 64, 0, 1);
        }
        #pragma unroll
        for (int nt = 0; nt < 8; ++nt) {
            v8f acc = {0, 0, 0, 0, 0, 0, 0, 0};
            acc = wmma_bf16(g0, load_b(w1 + nt * 512, lane), acc);
            acc = wmma_bf16(g1, load_b(w1 + (8 + nt) * 512, lane), acc);
            int col = nt * 16 + colq;
            store_acc(stg, 128, lane, col, acc, Ob1[n * 128 + col], true, true);
        }
        lds_fence();
        v8f acc = {0, 0, 0, 0, 0, 0, 0, 0};
        #pragma unroll
        for (int kc = 0; kc < 4; ++kc)
            acc = wmma_bf16(load_a(stg, lane, kc, 128), load_b(w2 + kc * 512, lane), acc);
        if (colq < 3) {
            float bv = Ob2[n * 3 + colq];
            int rb = (lane < 16) ? 0 : 8;
            #pragma unroll
            for (int r = 0; r < 8; ++r)
                sm[(rb + r) * 4 + colq] = acc[r] + bv;
        }
        lds_fence();
        if (lane < 16 && (row0 + lane) < rows) {
            float l0 = sm[lane * 4 + 0];
            float l1 = sm[lane * 4 + 1];
            float l2 = sm[lane * 4 + 2];
            float m  = fmaxf(l0, fmaxf(l1, l2));
            float e0 = __expf(l0 - m), e1 = __expf(l1 - m), e2 = __expf(l2 - m);
            float s  = 1.0f / (e0 + e1 + e2);
            int o = (row0 + lane) * 162 + n * 3;
            out[o + 0] = e0 * s;
            out[o + 1] = e1 * s;
            out[o + 2] = e2 * s;
        }
    }
}

extern "C" void kernel_launch(void* const* d_in, const int* in_sizes, int n_in,
                              void* d_out, int out_size, void* d_ws, size_t ws_size,
                              hipStream_t stream)
{
    const float* x   = (const float*)d_in[0];
    const float* Cw1 = (const float*)d_in[1];
    const float* Cb1 = (const float*)d_in[2];
    const float* Cw2 = (const float*)d_in[3];
    const float* Cb2 = (const float*)d_in[4];
    const float* Nw1 = (const float*)d_in[5];
    const float* Nb1 = (const float*)d_in[6];
    const float* Nw2 = (const float*)d_in[7];
    const float* Nb2 = (const float*)d_in[8];
    const float* Fw1 = (const float*)d_in[9];
    const float* Fb1 = (const float*)d_in[10];
    const float* Fw2 = (const float*)d_in[11];
    const float* Fb2 = (const float*)d_in[12];
    const float* Ow1 = (const float*)d_in[13];
    const float* Ob1 = (const float*)d_in[14];
    const float* Ow2 = (const float*)d_in[15];
    const float* Ob2 = (const float*)d_in[16];
    float*  out = (float*)d_out;
    __bf16* ws  = (__bf16*)d_ws;

    int rows = in_sizes[0] / 58;

    // ---- weight prep: f32 -> zero-padded bf16, B-operand lane layout --------
    auto prep = [&](const float* src, int offElems, int Ksrc, int Nsrc,
                    int KC, int NT, int nb) {
        int total = KC * NT * 512 * nb;
        prep_weights<<<(total + 255) / 256, 256, 0, stream>>>(
            src, ws + offElems, Ksrc, Nsrc, KC, NT, nb);
    };
    prep(Cw1, OFF_CW1,   4, 128, 1, 8,  1);
    prep(Cw2, OFF_CW2, 128,   3, 4, 1,  1);
    prep(Nw1, OFF_NW1,  54, 128, 2, 8,  1);
    prep(Nw2, OFF_NW2, 128,  54, 4, 4,  1);
    prep(Fw1, OFF_FW1,  57, 128, 2, 8,  1);
    prep(Fw2, OFF_FW2, 128,  57, 4, 4,  1);
    prep(Ow1, OFF_OW1,  57, 128, 2, 8, 54);
    prep(Ow2, OFF_OW2, 128,   3, 4, 1, 54);

    // ---- main fused kernel: 8 independent 16-row wave tiles per block -------
    int blocks = (rows + 127) / 128;
    if (blocks > 0) {
        dualnet_wmma<<<blocks, 256, 0, stream>>>(
            x, ws, Cb1, Cb2, Nb1, Nb2, Fb1, Fb2, Ob1, Ob2, out, rows);
    }
    (void)n_in; (void)out_size; (void)ws_size;
}
